// RRN_72507637891255
// MI455X (gfx1250) — compile-verified, matrix operand order checked by reading, not run on hardware
//
#include <hip/hip_runtime.h>
#include <hip/hip_bf16.h>
#include <math.h>

// ---------------------------------------------------------------------------
// RRN (recurrent relational network) — persistent per-batch kernel for gfx1250
//   grid = 64 blocks (one per batch element), block = 512 threads (16 wave32)
//   all recurrent state lives in LDS/registers; matmuls on v_wmma_f32_16x16x32_f16
//   v4: run-length-reduced message atomics (2 per lane per n-tile instead of 8),
//       vectorized float4 LDS copies, H-copy folded into readout L0 (one fewer
//       barrier per iteration).
// ---------------------------------------------------------------------------

typedef __attribute__((ext_vector_type(16))) _Float16 v16h;
typedef __attribute__((ext_vector_type(8)))  _Float16 v8h;
typedef __attribute__((ext_vector_type(8)))  float    v8f;

#define DEVINL __device__ __forceinline__

constexpr int NNODE   = 81;
constexpr int HD      = 96;
constexpr int NEDGE   = 20;
constexpr int NE      = NNODE * NEDGE;        // 1620 edges
constexpr int NEPAD   = 1632;                 // padded to 102 full tiles
constexpr int NTILE_E = NEPAD / 16;           // 102
constexpr int BLOCK   = 512;
constexpr int NWAVES  = BLOCK / 32;           // 16

// ---- LDS layout (bytes; all offsets 16B aligned) --------------------------
constexpr int OFF_EDGE = 0;                   // 1632 u8 edge targets
constexpr int OFF_BIAS = 1632;                // 1840 f32 bias pool = 7360
constexpr int OFF_HN   = 8992;                // 96*96 f32 (rows 81..95 bounded pad)
constexpr int OFF_X    = OFF_HN  + 36864;     // 96*96 f32 (rows 81..95 zero)
constexpr int OFF_MS   = OFF_X   + 36864;     // 96*96 f32 (rows 81.. = pad sinks)
constexpr int OFF_AI   = OFF_MS  + 36864;     // 96*96 f32
constexpr int OFF_AJ   = OFF_AI  + 36864;     // 96*96 f32
constexpr int OFF_FW1  = OFF_AJ  + 36864;     // 96*96 f16 hot msg weight
constexpr int OFF_FW2  = OFF_FW1 + 18432;     // 96*96 f16
constexpr int OFF_RW2  = OFF_FW2 + 18432;     // 16*96 f16 (r_w2 zero-padded)
constexpr int OFF_U    = OFF_RW2 + 3072;      // 73728B union scratch:
                                              //   msg phase: 16 waves * 16*96 f16 stage
                                              //   mlp phase: S0 + S1 (96*96 f32 each)
constexpr int SMEM_BYTES = OFF_U + 73728;     // 306976 <= 327680 (320KB)

// bias pool (float indices)
enum { IB0=0, IB1=96, IB2=192, FB0=288, FB1=384, FB2=480,
       GB0=576, GB1=672, GB2=768, RB0=864, RB1=960, RB2=1056,
       LBI=1072, LBH=1456, BIAS_TOTAL=1840 };

struct Params {
  const int*   grids;   const float* embed_W;
  const float *in_w0,*in_b0,*in_w1,*in_b1,*in_w2,*in_b2;
  const float *f_w0,*f_b0,*f_w1,*f_b1,*f_w2,*f_b2;
  const float *g_w0,*g_b0,*g_w1,*g_b1,*g_w2,*g_b2;
  const float *r_w0,*r_b0,*r_w1,*r_b1,*r_w2,*r_b2;
  const float *lstm_wi,*lstm_wh,*lstm_bi,*lstm_bh;
  const float *c0;  const int* iters;
  float* out;
};

// --------------------------- WMMA helpers ----------------------------------
DEVINL v8f wmma16(v16h a, v16h b, v8f c) {
  // D(16x16,f32) = A(16x32,f16) * B(32x16,f16) + C
  return __builtin_amdgcn_wmma_f32_16x16x32_f16(
      false, a, false, b, (short)0, c, false, false);
}

DEVINL v8f vsplat(float v) {
  v8f r;
  #pragma unroll
  for (int i = 0; i < 8; ++i) r[i] = v;
  return r;
}

// Branch-free A/B tile gathers. A tile: 16x32 f16 from f32 row-major source.
// Same gather loads B(32x16) tiles from weight W (out x in) row-major because
// B[k][n] = W[n][k] and the assumed B layout mirrors the A layout on W rows.
DEVINL v16h ldA32(const float* __restrict__ src, int ld,
                  int row0, int kOff, int lane) {
  const int r  = row0 + (lane & 15);
  const int kb = (lane & 16) ? 8 : 0;
  const float* p = src + (size_t)r * ld + kOff + kb;
  const float4 f0 = *reinterpret_cast<const float4*>(p);
  const float4 f1 = *reinterpret_cast<const float4*>(p + 4);
  const float4 f2 = *reinterpret_cast<const float4*>(p + 16);
  const float4 f3 = *reinterpret_cast<const float4*>(p + 20);
  v16h o;
  o[0]=(_Float16)f0.x;  o[1]=(_Float16)f0.y;  o[2]=(_Float16)f0.z;  o[3]=(_Float16)f0.w;
  o[4]=(_Float16)f1.x;  o[5]=(_Float16)f1.y;  o[6]=(_Float16)f1.z;  o[7]=(_Float16)f1.w;
  o[8]=(_Float16)f2.x;  o[9]=(_Float16)f2.y;  o[10]=(_Float16)f2.z; o[11]=(_Float16)f2.w;
  o[12]=(_Float16)f3.x; o[13]=(_Float16)f3.y; o[14]=(_Float16)f3.z; o[15]=(_Float16)f3.w;
  return o;
}

DEVINL v16h ldA16(const _Float16* __restrict__ src, int ld,
                  int row0, int kOff, int lane) {
  const int r  = row0 + (lane & 15);
  const int kb = (lane & 16) ? 8 : 0;
  const _Float16* p = src + (size_t)r * ld + kOff + kb;
  const v8h lo = *reinterpret_cast<const v8h*>(p);
  const v8h hi = *reinterpret_cast<const v8h*>(p + 16);
  v16h o;
  #pragma unroll
  for (int h = 0; h < 8; ++h) { o[h] = lo[h]; o[8 + h] = hi[h]; }
  return o;
}

// acc += A(16 rows of aSrc) * W(wRow0..wRow0+15)^T over K=96 (3 chunks)
DEVINL v8f gemmK96(v8f acc, const float* aSrc, int aRow0,
                   const float* W, int wLd, int wRow0, int lane) {
  #pragma unroll
  for (int kc = 0; kc < 3; ++kc) {
    v16h a  = ldA32(aSrc, HD, aRow0, kc * 32, lane);
    v16h bt = ldA32(W, wLd, wRow0, kc * 32, lane);
    acc = wmma16(a, bt, acc);
  }
  return acc;
}

// D-tile store (unconditional): lane holds column n, rows r + (lane16 ? 8 : 0)
DEVINL void storeDu(float* dst, int ld, int row0, int col0,
                    const v8f& acc, bool relu, int lane) {
  const int n  = col0 + (lane & 15);
  const int rb = row0 + ((lane & 16) ? 8 : 0);
  #pragma unroll
  for (int r = 0; r < 8; ++r)
    dst[(rb + r) * ld + n] = relu ? fmaxf(acc[r], 0.f) : acc[r];
}

DEVINL float sigm(float x) { return 1.f / (1.f + __expf(-x)); }

// ---------------------------------------------------------------------------
__global__ __launch_bounds__(BLOCK, 1)
void rrn_persistent_kernel(Params p) {
  extern __shared__ char smem[];
  unsigned char* edgeL = (unsigned char*)(smem + OFF_EDGE);
  float*    bias  = (float*)   (smem + OFF_BIAS);
  float*    Hn    = (float*)   (smem + OFF_HN);
  float*    Xb    = (float*)   (smem + OFF_X);
  float*    Ms    = (float*)   (smem + OFF_MS);
  float*    Ai    = (float*)   (smem + OFF_AI);
  float*    Aj    = (float*)   (smem + OFF_AJ);
  _Float16* fW1h  = (_Float16*)(smem + OFF_FW1);
  _Float16* fW2h  = (_Float16*)(smem + OFF_FW2);
  _Float16* rW2h  = (_Float16*)(smem + OFF_RW2);
  float*    S0    = (float*)   (smem + OFF_U);
  float*    S1    = (float*)   (smem + OFF_U + 36864);
  _Float16* stage = (_Float16*)(smem + OFF_U);   // msg-phase only (overlaps S0/S1)

  const int bId  = blockIdx.x;
  const int tid  = threadIdx.x;
  const int lane = tid & 31;
  const int wave = tid >> 5;

  // ---- one-time setup -----------------------------------------------------
  // sudoku edge lists (sorted ascending == reference determine_edges(3,3))
  if (tid < NNODE) {
    const int r = tid / 9, c = tid % 9;
    int cnt = 0;
    for (int m = 0; m < 81; ++m) {
      const int r2 = m / 9, c2 = m % 9;
      const bool nb = (m != tid) &&
                      (r2 == r || c2 == c || (r2 / 3 == r / 3 && c2 / 3 == c / 3));
      if (nb) edgeL[tid * NEDGE + (cnt++)] = (unsigned char)m;
    }
  }
  if (tid < NEPAD - NE) edgeL[NE + tid] = 0;           // pad edges -> node 0
  if (tid < 96) {
    bias[IB0+tid]=p.in_b0[tid]; bias[IB1+tid]=p.in_b1[tid]; bias[IB2+tid]=p.in_b2[tid];
    bias[FB0+tid]=p.f_b0[tid];  bias[FB1+tid]=p.f_b1[tid];  bias[FB2+tid]=p.f_b2[tid];
    bias[GB0+tid]=p.g_b0[tid];  bias[GB1+tid]=p.g_b1[tid];  bias[GB2+tid]=p.g_b2[tid];
    bias[RB0+tid]=p.r_b0[tid];  bias[RB1+tid]=p.r_b1[tid];
  }
  if (tid < 16)  bias[RB2+tid] = (tid < 9) ? p.r_b2[tid] : 0.f;
  if (tid < 384) { bias[LBI+tid] = p.lstm_bi[tid]; bias[LBH+tid] = p.lstm_bh[tid]; }
  for (int i = tid; i < 96 * 96; i += BLOCK) {
    fW1h[i] = (_Float16)p.f_w1[i];
    fW2h[i] = (_Float16)p.f_w2[i];
  }
  for (int i = tid; i < 16 * 96; i += BLOCK) {         // r_w2 zero-padded 9->16
    const int row = i / 96, col = i % 96;
    rW2h[i] = (row < 9) ? (_Float16)p.r_w2[row * 96 + col] : (_Float16)0.f;
  }
  for (int i = NNODE * HD + tid; i < 96 * HD; i += BLOCK) {  // zero pad rows
    Hn[i] = 0.f; Xb[i] = 0.f;
  }

  // LSTM cell state lives in registers: slot s -> job = wave + s*16 (fixed map)
  float creg[3][8];
  #pragma unroll
  for (int s = 0; s < 3; ++s) {
    const int job = wave + s * NWAVES;
    if (job < 36) {
      const int mt = job / 6, ht = job % 6;
      const int ncol = ht * 16 + (lane & 15);
      const int rbm  = mt * 16 + ((lane & 16) ? 8 : 0);
      #pragma unroll
      for (int r = 0; r < 8; ++r) {
        const int node = rbm + r;
        creg[s][r] = (node < NNODE)
            ? p.c0[((size_t)bId * NNODE + node) * HD + ncol] : 0.f;
      }
    } else {
      #pragma unroll
      for (int r = 0; r < 8; ++r) creg[s][r] = 0.f;
    }
  }
  __syncthreads();

  // ---- input MLP: X = mlp3(embed[grids]) (scalar fp32, one-time, tiny) ----
  for (int idx = tid; idx < NNODE * HD; idx += BLOCK) {
    const int n = idx / HD, h = idx % HD;
    const int g = p.grids[bId * NNODE + n];
    const float* ev = p.embed_W + g * 16;
    const float* wv = p.in_w0 + h * 16;
    float s = bias[IB0 + h];
    #pragma unroll
    for (int k = 0; k < 16; ++k) s += ev[k] * wv[k];
    S0[idx] = fmaxf(s, 0.f);
  }
  __syncthreads();
  for (int idx = tid; idx < NNODE * HD; idx += BLOCK) {
    const int n = idx / HD, h = idx % HD;
    const float* a = S0 + n * HD;
    const float* w = p.in_w1 + h * HD;
    float s = bias[IB1 + h];
    for (int k = 0; k < HD; ++k) s += a[k] * w[k];
    S1[idx] = fmaxf(s, 0.f);
  }
  __syncthreads();
  for (int idx = tid; idx < NNODE * HD; idx += BLOCK) {
    const int n = idx / HD, h = idx % HD;
    const float* a = S1 + n * HD;
    const float* w = p.in_w2 + h * HD;
    float s = bias[IB2 + h];
    for (int k = 0; k < HD; ++k) s += a[k] * w[k];
    Xb[idx] = s;            // final layer: no relu
    Hn[idx] = s;            // H0 = stop_gradient(X) = X
  }
  __syncthreads();

  const int ITERS = p.iters[0];

  for (int it = 0; it < ITERS; ++it) {
    // == phase 1: layer-0 factorization  Ai = W0a@Hn, Aj = W0b@Hn ===========
    for (int job = wave; job < 72; job += NWAVES) {
      const int which = job / 36;               // 0 -> Ai, 1 -> Aj
      const int rem = job % 36, mt = rem / 6, nt = rem % 6;
      const float* W = p.f_w0 + (which ? 96 : 0);   // (96 x 192) row-major
      v8f acc = vsplat(0.f);
      acc = gemmK96(acc, Hn, mt * 16, W, 192, nt * 16, lane);
      storeDu(which ? Aj : Ai, HD, mt * 16, nt * 16, acc, false, lane);
    }
    {   // vectorized Ms zero (float4)
      float4* mz = (float4*)Ms;
      for (int i = tid; i < (96 * HD) / 4; i += BLOCK)
        mz[i] = float4{0.f, 0.f, 0.f, 0.f};
    }
    __syncthreads();

    // == phase 2: per-edge message MLP (layers 1-2 on WMMA) =================
    {
      _Float16* stg = stage + wave * (16 * HD);
      const int m  = lane & 15;
      const int kb = (lane & 16) ? 8 : 0;
      const int rb = (lane & 16) ? 8 : 0;
      for (int tile = wave; tile < NTILE_E; tile += NWAVES) {
        const int e  = tile * 16 + m;           // < 1632; pads -> Ai row 81
        const int ni = e / NEDGE;
        const int nj = edgeL[e];
        const float* pi = Ai + ni * HD;
        const float* pj = Aj + nj * HD;
        v16h A0[3];
        #pragma unroll
        for (int kc = 0; kc < 3; ++kc) {
          #pragma unroll
          for (int g = 0; g < 2; ++g) {
            const int k0 = kc * 32 + kb + g * 16;
            const float4 i0 = *reinterpret_cast<const float4*>(pi + k0);
            const float4 i1 = *reinterpret_cast<const float4*>(pi + k0 + 4);
            const float4 j0 = *reinterpret_cast<const float4*>(pj + k0);
            const float4 j1 = *reinterpret_cast<const float4*>(pj + k0 + 4);
            const float4 b0 = *reinterpret_cast<const float4*>(&bias[FB0 + k0]);
            const float4 b1 = *reinterpret_cast<const float4*>(&bias[FB0 + k0 + 4]);
            A0[kc][g*8+0] = (_Float16)fmaxf(i0.x + j0.x + b0.x, 0.f);
            A0[kc][g*8+1] = (_Float16)fmaxf(i0.y + j0.y + b0.y, 0.f);
            A0[kc][g*8+2] = (_Float16)fmaxf(i0.z + j0.z + b0.z, 0.f);
            A0[kc][g*8+3] = (_Float16)fmaxf(i0.w + j0.w + b0.w, 0.f);
            A0[kc][g*8+4] = (_Float16)fmaxf(i1.x + j1.x + b1.x, 0.f);
            A0[kc][g*8+5] = (_Float16)fmaxf(i1.y + j1.y + b1.y, 0.f);
            A0[kc][g*8+6] = (_Float16)fmaxf(i1.z + j1.z + b1.z, 0.f);
            A0[kc][g*8+7] = (_Float16)fmaxf(i1.w + j1.w + b1.w, 0.f);
          }
        }
        // L1: act1 = relu(A0 @ fW1^T + b1) -> f16 stage (re-layout D -> A)
        for (int nt = 0; nt < 6; ++nt) {
          const int n = nt * 16 + m;
          v8f acc = vsplat(bias[FB1 + n]);
          #pragma unroll
          for (int kc = 0; kc < 3; ++kc)
            acc = wmma16(A0[kc], ldA16(fW1h, HD, nt * 16, kc * 32, lane), acc);
          #pragma unroll
          for (int r = 0; r < 8; ++r)
            stg[(rb + r) * HD + n] = (_Float16)fmaxf(acc[r], 0.f);
        }
        v16h A1[3];
        #pragma unroll
        for (int kc = 0; kc < 3; ++kc)
          A1[kc] = ldA16(stg, HD, 0, kc * 32, lane);
        // L2: msg = A1 @ fW2^T + b2 (no relu). A lane's 8 D-rows are 8
        // consecutive edges spanning at most 2 nodes -> pre-reduce into two
        // partial sums, 2 unconditional LDS atomics (pads land in pad rows).
        const int e0    = tile * 16 + rb;       // first edge of lane's rows
        const int nodeA = e0 / NEDGE;           // <= 81
        const int split = nodeA * NEDGE + NEDGE - e0;   // rows in nodeA (1..20)
        for (int nt = 0; nt < 6; ++nt) {
          const int n = nt * 16 + m;
          v8f acc = vsplat(bias[FB2 + n]);
          #pragma unroll
          for (int kc = 0; kc < 3; ++kc)
            acc = wmma16(A1[kc], ldA16(fW2h, HD, nt * 16, kc * 32, lane), acc);
          float sA = 0.f, sB = 0.f;
          #pragma unroll
          for (int r = 0; r < 8; ++r) {
            sA += (r < split) ? acc[r] : 0.f;
            sB += (r < split) ? 0.f : acc[r];
          }
          atomicAdd(&Ms[nodeA * HD + n], sA);
          atomicAdd(&Ms[(nodeA + 1) * HD + n], sB);   // nodeA+1 <= 82 < 96
        }
      }
    }
    __syncthreads();

    // == phase 3: g-MLP  gin = mlp3(concat(X, M)) ===========================
    for (int job = wave; job < 36; job += NWAVES) {          // L0
      const int mt = job / 6, nt = job % 6;
      v8f acc = vsplat(bias[GB0 + nt * 16 + (lane & 15)]);
      acc = gemmK96(acc, Xb, mt * 16, p.g_w0,      192, nt * 16, lane);
      acc = gemmK96(acc, Ms, mt * 16, p.g_w0 + 96, 192, nt * 16, lane);
      storeDu(S0, HD, mt * 16, nt * 16, acc, true, lane);
    }
    __syncthreads();
    for (int job = wave; job < 36; job += NWAVES) {          // L1
      const int mt = job / 6, nt = job % 6;
      v8f acc = vsplat(bias[GB1 + nt * 16 + (lane & 15)]);
      acc = gemmK96(acc, S0, mt * 16, p.g_w1, HD, nt * 16, lane);
      storeDu(S1, HD, mt * 16, nt * 16, acc, true, lane);
    }
    __syncthreads();
    for (int job = wave; job < 36; job += NWAVES) {          // L2 -> gin (S0)
      const int mt = job / 6, nt = job % 6;
      v8f acc = vsplat(bias[GB2 + nt * 16 + (lane & 15)]);
      acc = gemmK96(acc, S1, mt * 16, p.g_w2, HD, nt * 16, lane);
      storeDu(S0, HD, mt * 16, nt * 16, acc, false, lane);
    }
    __syncthreads();

    // == phase 4: LSTM cell (gates on WMMA, state in registers) =============
    #pragma unroll
    for (int s = 0; s < 3; ++s) {
      const int job = wave + s * NWAVES;        // wave-uniform condition
      if (job >= 36) break;
      const int mt = job / 6, ht = job % 6;
      v16h Ag[3], Ah[3];
      #pragma unroll
      for (int kc = 0; kc < 3; ++kc) {
        Ag[kc] = ldA32(S0, HD, mt * 16, kc * 32, lane);  // gin
        Ah[kc] = ldA32(Hn, HD, mt * 16, kc * 32, lane);  // hprev
      }
      const int ncol = ht * 16 + (lane & 15);
      v8f gacc[4];
      #pragma unroll
      for (int gi = 0; gi < 4; ++gi) {
        const int wrow = gi * 96 + ht * 16;
        gacc[gi] = vsplat(bias[LBI + gi * 96 + ncol] + bias[LBH + gi * 96 + ncol]);
        #pragma unroll
        for (int kc = 0; kc < 3; ++kc)
          gacc[gi] = wmma16(Ag[kc], ldA32(p.lstm_wi, HD, wrow, kc * 32, lane), gacc[gi]);
        #pragma unroll
        for (int kc = 0; kc < 3; ++kc)
          gacc[gi] = wmma16(Ah[kc], ldA32(p.lstm_wh, HD, wrow, kc * 32, lane), gacc[gi]);
      }
      const int rbm = mt * 16 + ((lane & 16) ? 8 : 0);
      #pragma unroll
      for (int r = 0; r < 8; ++r) {
        const float cn = sigm(gacc[1][r]) * creg[s][r]
                       + sigm(gacc[0][r]) * tanhf(gacc[2][r]);
        creg[s][r] = cn;
        S1[(rbm + r) * HD + ncol] = sigm(gacc[3][r]) * tanhf(cn);  // Hnew
      }
    }
    __syncthreads();

    // == phase 5: readout r-MLP on Hnew (S1); Hn <- S1 folded in ============
    for (int job = wave; job < 36; job += NWAVES) {          // L0 reads S1
      const int mt = job / 6, nt = job % 6;
      v8f acc = vsplat(bias[RB0 + nt * 16 + (lane & 15)]);
      acc = gemmK96(acc, S1, mt * 16, p.r_w0, HD, nt * 16, lane);
      storeDu(S0, HD, mt * 16, nt * 16, acc, true, lane);
    }
    {   // Hn <- S1 (hprev for next iteration), vectorized float4
      const float4* src4 = (const float4*)S1;
      float4*       dst4 = (float4*)Hn;
      for (int i = tid; i < (96 * HD) / 4; i += BLOCK) dst4[i] = src4[i];
    }
    __syncthreads();
    for (int job = wave; job < 36; job += NWAVES) {          // L1
      const int mt = job / 6, nt = job % 6;
      v8f acc = vsplat(bias[RB1 + nt * 16 + (lane & 15)]);
      acc = gemmK96(acc, S0, mt * 16, p.r_w1, HD, nt * 16, lane);
      storeDu(S1, HD, mt * 16, nt * 16, acc, true, lane);
    }
    __syncthreads();
    for (int mt = wave; mt < 6; mt += NWAVES) {              // L2 (out dim 9)
      const int n = lane & 15;
      v8f acc = vsplat(bias[RB2 + n]);                       // padded cols -> 0
      #pragma unroll
      for (int kc = 0; kc < 3; ++kc)
        acc = wmma16(ldA32(S1, HD, mt * 16, kc * 32, lane),
                     ldA16(rW2h, HD, 0, kc * 32, lane), acc);
      const int rbm = mt * 16 + ((lane & 16) ? 8 : 0);
      if (n < 9) {                                           // real global write
        #pragma unroll
        for (int r = 0; r < 8; ++r) {
          const int node = rbm + r;
          if (node < NNODE)
            p.out[(((size_t)it * 64 + bId) * NNODE + node) * 9 + n] = acc[r];
        }
      }
    }
    __syncthreads();
  }
}

// ---------------------------------------------------------------------------
extern "C" void kernel_launch(void* const* d_in, const int* in_sizes, int n_in,
                              void* d_out, int out_size, void* d_ws, size_t ws_size,
                              hipStream_t stream) {
  (void)in_sizes; (void)n_in; (void)out_size; (void)d_ws; (void)ws_size;
  Params p;
  p.grids   = (const int*)  d_in[0];
  p.embed_W = (const float*)d_in[1];
  p.in_w0 = (const float*)d_in[2];  p.in_b0 = (const float*)d_in[3];
  p.in_w1 = (const float*)d_in[4];  p.in_b1 = (const float*)d_in[5];
  p.in_w2 = (const float*)d_in[6];  p.in_b2 = (const float*)d_in[7];
  p.f_w0  = (const float*)d_in[8];  p.f_b0  = (const float*)d_in[9];
  p.f_w1  = (const float*)d_in[10]; p.f_b1  = (const float*)d_in[11];
  p.f_w2  = (const float*)d_in[12]; p.f_b2  = (const float*)d_in[13];
  p.g_w0  = (const float*)d_in[14]; p.g_b0  = (const float*)d_in[15];
  p.g_w1  = (const float*)d_in[16]; p.g_b1  = (const float*)d_in[17];
  p.g_w2  = (const float*)d_in[18]; p.g_b2  = (const float*)d_in[19];
  p.r_w0  = (const float*)d_in[20]; p.r_b0  = (const float*)d_in[21];
  p.r_w1  = (const float*)d_in[22]; p.r_b1  = (const float*)d_in[23];
  p.r_w2  = (const float*)d_in[24]; p.r_b2  = (const float*)d_in[25];
  p.lstm_wi = (const float*)d_in[26]; p.lstm_wh = (const float*)d_in[27];
  p.lstm_bi = (const float*)d_in[28]; p.lstm_bh = (const float*)d_in[29];
  p.c0    = (const float*)d_in[30];
  p.iters = (const int*)  d_in[31];
  p.out   = (float*)d_out;

  (void)hipFuncSetAttribute((const void*)rrn_persistent_kernel,
                            hipFuncAttributeMaxDynamicSharedMemorySize,
                            SMEM_BYTES);
  rrn_persistent_kernel<<<dim3(64), dim3(BLOCK), SMEM_BYTES, stream>>>(p);
}
